// GPUOptimizedLiquidLayer_50929722196025
// MI455X (gfx1250) — matrix operand order repaired and checked by reading, not compiled, not run on hardware
//
#include <hip/hip_runtime.h>
#include <hip/hip_bf16.h>
#include <math.h>

// Problem constants (match reference)
#define BB 64
#define SS 512
#define DD 1024
#define HH 1024
#define KK 4
#define HB_LD 1032   // padded bf16 row stride for h in LDS (4-bank skew, 16B aligned)

typedef __bf16 bf16_t;
typedef bf16_t v16bf __attribute__((ext_vector_type(16)));
typedef bf16_t v8bf  __attribute__((ext_vector_type(8)));
typedef float  v8f   __attribute__((ext_vector_type(8)));

// ---------------------------------------------------------------------------
// Fragment loaders (CDNA5 WMMA 16x16x32 bf16 layouts, wave32)
// A 16x32 (MxK): row = lane&15, k-group = (lane>>4)*8:
//   elems 0..7  -> k = k0 + kg + e ; elems 8..15 -> k = k0 + 16 + kg + (e-8)
// B 32x16 (KxN), B[k][n] = W[n][k]: row n = n0+(lane&15), same k order as A.
// ---------------------------------------------------------------------------
__device__ __forceinline__ v16bf load_a_f32(const float* __restrict__ A,
                                            int m0, long ld, int k0, int lane) {
  const int m  = m0 + (lane & 15);
  const int kg = (lane >> 4) << 3;
  const float* p = A + (long)m * ld + (k0 + kg);
  v8f a0 = *(const v8f*)(p);
  v8f a1 = *(const v8f*)(p + 16);
  v16bf r;
#pragma unroll
  for (int i = 0; i < 8; ++i) {
    r[i]     = (bf16_t)a0[i];
    r[8 + i] = (bf16_t)a1[i];
  }
  return r;
}

__device__ __forceinline__ v16bf load_b_bf16(const bf16_t* __restrict__ W,
                                             int n0, int k0, int lane) {
  const int n  = n0 + (lane & 15);
  const int kg = (lane >> 4) << 3;
  const bf16_t* p = W + (long)n * HH + (k0 + kg);
  v8bf b0 = *(const v8bf*)(p);
  v8bf b1 = *(const v8bf*)(p + 16);
  v16bf r;
#pragma unroll
  for (int i = 0; i < 8; ++i) {
    r[i]     = b0[i];
    r[8 + i] = b1[i];
  }
  return r;
}

// ---------------------------------------------------------------------------
// Small helper kernels
// ---------------------------------------------------------------------------
__global__ void cvt_f32_to_bf16(const float* __restrict__ src,
                                bf16_t* __restrict__ dst, int n) {
  int i = blockIdx.x * blockDim.x + threadIdx.x;
  if (i < n) dst[i] = (bf16_t)src[i];
}

__global__ void zero_f32(float* __restrict__ p, int n) {
  int i = blockIdx.x * blockDim.x + threadIdx.x;
  if (i < n) p[i] = 0.0f;
}

// alpha[b] = exp(-DT / sum_k tau_k * softmax_k(c[b]*mw[k] + mb[k]))
__global__ void liquid_alpha(const float* __restrict__ cx,
                             const float* __restrict__ mw,
                             const float* __restrict__ mb,
                             const float* __restrict__ tau,
                             float* __restrict__ alpha) {
  int b = threadIdx.x;
  if (b >= BB) return;
  float c = cx[b];
  float l[KK];
  float mx = -1e30f;
#pragma unroll
  for (int k = 0; k < KK; ++k) {
    l[k] = c * mw[k] + mb[k];
    mx = fmaxf(mx, l[k]);
  }
  float se = 0.f, tw = 0.f;
#pragma unroll
  for (int k = 0; k < KK; ++k) {
    float e = __expf(l[k] - mx);
    se += e;
    tw += e * tau[k];
  }
  alpha[b] = __expf(-0.05f / (tw / se));
}

// h_final[b, n] = out[b, S-1, n]
__global__ void copy_final(const float* __restrict__ srcLastT,
                           float* __restrict__ dst) {
  int i = blockIdx.x * blockDim.x + threadIdx.x;   // 0 .. B*H-1
  int b = i >> 10;
  int n = i & (HH - 1);
  dst[i] = srcLastT[(long)b * SS * HH + n];
}

// ---------------------------------------------------------------------------
// Phase 1: U = X @ W_in^T + bias, written straight into d_out ([B*S, H]).
// Wave computes a 16x64 tile (A reused over 4 N-subtiles), 128 v_wmma each.
// grid = (256, 16), block = 256 (8 waves).
// ---------------------------------------------------------------------------
__global__ void __launch_bounds__(256)
liquid_input_proj(const float* __restrict__ X, const bf16_t* __restrict__ Win,
                  const float* __restrict__ bias, float* __restrict__ U) {
  const int lane = threadIdx.x & 31;
  const int wave = threadIdx.x >> 5;
  const int m0 = (blockIdx.x * 8 + wave) * 16;
  const int n0 = blockIdx.y * 64;

  v8f acc[4];
#pragma unroll
  for (int j = 0; j < 4; ++j)
#pragma unroll
    for (int i = 0; i < 8; ++i) acc[j][i] = 0.0f;

  for (int k0 = 0; k0 < DD; k0 += 32) {
    v16bf a = load_a_f32(X, m0, DD, k0, lane);
#pragma unroll
    for (int j = 0; j < 4; ++j) {
      v16bf b = load_b_bf16(Win, n0 + 16 * j, k0, lane);
      acc[j] = __builtin_amdgcn_wmma_f32_16x16x32_bf16(
          false, a, false, b, (short)0, acc[j], false, false);
    }
  }

  const int mrow = m0 + ((lane >> 4) << 3);
  const int nc   = n0 + (lane & 15);
#pragma unroll
  for (int j = 0; j < 4; ++j) {
    float bj = bias[nc + 16 * j];
#pragma unroll
    for (int r = 0; r < 8; ++r)
      U[(long)(mrow + r) * HH + nc + 16 * j] = acc[j][r] + bj;
  }
}

// ---------------------------------------------------------------------------
// Phase 2: persistent scan. The recurrence is batch-row independent, so each
// workgroup owns a 16-row batch tile and runs ALL 512 steps with only
// workgroup barriers between them (no per-step launches).
//   - block = 1024 threads (32 waves); wave w owns N-tiles 2w, 2w+1.
//   - h (bf16) lives in LDS (16 x HB_LD), read as A-fragments via ds_load_b128.
//   - out (U -> h, in place) and f32 state hs are touched by exactly one
//     owning lane for the whole scan => no cross-CU global coherence needed.
//   - U for step t is prefetched (global_prefetch_b8) before the K-loop.
// grid = 4 blocks.
// ---------------------------------------------------------------------------
__global__ void __launch_bounds__(1024)
liquid_scan(const bf16_t* __restrict__ Wrec,
            const float* __restrict__ alpha,
            float* __restrict__ hs,        // [B][H] f32 state, pre-zeroed
            float* __restrict__ out) {     // [B][S][H], holds U on entry
  __shared__ bf16_t hb[16 * HB_LD];        // ~33 KB

  const int tid  = threadIdx.x;
  const int lane = tid & 31;
  const int wave = tid >> 5;
  const int m0   = blockIdx.x * 16;        // batch rows [m0, m0+16)

  for (int i = tid; i < 16 * HB_LD; i += 1024) hb[i] = (bf16_t)0.0f;
  __syncthreads();

  const int rbase = (lane >> 4) << 3;      // C-layout row base for this lane
  const int ncl   = lane & 15;             // C-layout column within tile
  const int n0a   = wave * 32;             // this wave's two N-tiles
  const int n0b   = n0a + 16;

  float al[8];
#pragma unroll
  for (int r = 0; r < 8; ++r) al[r] = alpha[m0 + rbase + r];

  for (int t = 0; t < SS; ++t) {
    // Prefetch this step's U values (epilogue operands) into cache.
#pragma unroll
    for (int r = 0; r < 8; ++r) {
      long g = ((long)(m0 + rbase + r) * SS + t) * HH + ncl;
      __builtin_prefetch(out + g + n0a, 0, 0);
      __builtin_prefetch(out + g + n0b, 0, 0);
    }

    v8f acc0, acc1;
#pragma unroll
    for (int i = 0; i < 8; ++i) { acc0[i] = 0.0f; acc1[i] = 0.0f; }

    for (int k0 = 0; k0 < HH; k0 += 32) {
      // A-fragment from LDS (two ds_load_b128, conflict-free via HB_LD skew)
      const bf16_t* pa = hb + (lane & 15) * HB_LD + k0 + ((lane >> 4) << 3);
      v8bf a0 = *(const v8bf*)(pa);
      v8bf a1 = *(const v8bf*)(pa + 16);
      v16bf a;
#pragma unroll
      for (int i = 0; i < 8; ++i) { a[i] = a0[i]; a[8 + i] = a1[i]; }

      v16bf b0 = load_b_bf16(Wrec, n0a, k0, lane);
      v16bf b1 = load_b_bf16(Wrec, n0b, k0, lane);
      acc0 = __builtin_amdgcn_wmma_f32_16x16x32_bf16(
          false, a, false, b0, (short)0, acc0, false, false);
      acc1 = __builtin_amdgcn_wmma_f32_16x16x32_bf16(
          false, a, false, b1, (short)0, acc1, false, false);
    }

    __syncthreads();   // all waves done reading hb for step t

#pragma unroll
    for (int r = 0; r < 8; ++r) {
      const int  ml = rbase + r;
      const int  mg = m0 + ml;
      const long gb = ((long)mg * SS + t) * HH + ncl;
      {
        const int n = n0a + ncl;
        float u  = out[gb + n0a];                       // U_t (lane-owned)
        float hp = hs[(long)mg * HH + n];               // f32 h (lane-owned)
        float hn = al[r] * hp + (1.0f - al[r]) * tanhf(acc0[r] + u);
        out[gb + n0a]         = hn;
        hs[(long)mg * HH + n] = hn;
        hb[ml * HB_LD + n]    = (bf16_t)hn;
      }
      {
        const int n = n0b + ncl;
        float u  = out[gb + n0b];
        float hp = hs[(long)mg * HH + n];
        float hn = al[r] * hp + (1.0f - al[r]) * tanhf(acc1[r] + u);
        out[gb + n0b]         = hn;
        hs[(long)mg * HH + n] = hn;
        hb[ml * HB_LD + n]    = (bf16_t)hn;
      }
    }
    __syncthreads();   // h updates visible before next step's A-fragment reads
  }
}

// ---------------------------------------------------------------------------
// Launch
// ---------------------------------------------------------------------------
extern "C" void kernel_launch(void* const* d_in, const int* in_sizes, int n_in,
                              void* d_out, int out_size, void* d_ws, size_t ws_size,
                              hipStream_t stream) {
  (void)in_sizes; (void)n_in; (void)out_size; (void)ws_size;

  const float* x    = (const float*)d_in[0];  // [B, S, D]
  const float* cx   = (const float*)d_in[1];  // [B, 1]
  const float* Wrec = (const float*)d_in[2];  // [H, H]
  const float* Win  = (const float*)d_in[3];  // [D, H]
  const float* bias = (const float*)d_in[4];  // [H]
  const float* tau  = (const float*)d_in[5];  // [K]
  const float* mw   = (const float*)d_in[6];  // [K, 1]
  const float* mb   = (const float*)d_in[7];  // [K]
  float* out = (float*)d_out;                 // [B,S,H] then [B,H]

  // Workspace (~4.5 MB): bf16 weights (L2-resident), alpha, f32 h state.
  bf16_t* wrec_bf = (bf16_t*)d_ws;                       // H*H bf16 = 2 MB
  bf16_t* win_bf  = wrec_bf + (size_t)HH * HH;           // D*H bf16 = 2 MB
  float*  alpha   = (float*)(win_bf + (size_t)DD * HH);  // 64 f32
  float*  hs      = alpha + BB;                          // B*H f32 (256 KB)

  cvt_f32_to_bf16<<<(HH * HH) / 256, 256, 0, stream>>>(Wrec, wrec_bf, HH * HH);
  cvt_f32_to_bf16<<<(DD * HH) / 256, 256, 0, stream>>>(Win, win_bf, DD * HH);
  zero_f32<<<(BB * HH) / 256, 256, 0, stream>>>(hs, BB * HH);
  liquid_alpha<<<1, 64, 0, stream>>>(cx, mw, mb, tau, alpha);

  // Phase 1: U (+bias) straight into d_out. [32768 x 1024] x [1024 x 1024].
  liquid_input_proj<<<dim3(256, 16), 256, 0, stream>>>(x, win_bf, bias, out);

  // Phase 2: persistent 512-step scan, 4 workgroups (one per 16 batch rows).
  liquid_scan<<<4, 1024, 0, stream>>>(wrec_bf, alpha, hs, out);

  // h_final tail.
  copy_final<<<(BB * HH) / 256, 256, 0, stream>>>(out + (long)(SS - 1) * HH,
                                                  out + (long)BB * SS * HH);
}
